// SpatiallyEmbeddedRNN_4818953306665
// MI455X (gfx1250) — matrix operand order repaired and checked by reading, not compiled, not run on hardware
//
#include <hip/hip_runtime.h>
#include <hip/hip_bf16.h>

typedef __attribute__((ext_vector_type(16))) _Float16 v16h;
typedef __attribute__((ext_vector_type(8)))  float    v8f;

#define CH   32
#define HWD  128
#define IMG  (HWD * HWD)                 // 16384
#define PLANE ((size_t)CH * IMG)         // 524288 per batch
#define NPIX  ((size_t)16 * PLANE)       // 8388608 per tensor
#define COLS 130                         // 128 + zero-pad halo columns

// ---------------------------------------------------------------------------
// Repack 7 weight tensors (OIHW, 32x32x3x3) into WMMA A-fragment layout (f16).
// Fragment id = (tensor*9 + tap)*2 + mtile. Each fragment = 32 lanes x 16
// halfs (32B/lane contiguous). Layout (16-bit A, 16x32, contiguous K order, same as B):
//   M = (lane&15) + mtile*16
//   K = h + (lane>=16 ? 16 : 0)
// tensors: 0 W_in_e(id) 1 W_e_e(relu) 2 W_i_e(-relu)
//          3 W_in_i(id) 4 W_e_i(relu) 5 W_i_i(-relu) 6 W_e_out(relu)
// ---------------------------------------------------------------------------
__global__ void repack_weights_kernel(
    const float* __restrict__ W_in_e, const float* __restrict__ W_e_e,
    const float* __restrict__ W_i_e,  const float* __restrict__ W_in_i,
    const float* __restrict__ W_e_i,  const float* __restrict__ W_i_i,
    const float* __restrict__ W_e_out, _Float16* __restrict__ ws) {
  int gid = blockIdx.x * blockDim.x + threadIdx.x;
  if (gid >= 126 * 512) return;
  int frag = gid >> 9;
  int rem  = gid & 511;
  int lane = rem >> 4;
  int h    = rem & 15;
  int tensor = frag / 18;
  int rem2   = frag % 18;
  int tap    = rem2 >> 1;
  int mtile  = rem2 & 1;
  int ky = tap / 3, kx = tap % 3;
  int M = (lane & 15) + mtile * 16;
  int K = h + ((lane >= 16) ? 16 : 0);

  const float* W;
  int mode; // 0=identity, 1=relu, 2=-relu
  switch (tensor) {
    case 0: W = W_in_e;  mode = 0; break;
    case 1: W = W_e_e;   mode = 1; break;
    case 2: W = W_i_e;   mode = 2; break;
    case 3: W = W_in_i;  mode = 0; break;
    case 4: W = W_e_i;   mode = 1; break;
    case 5: W = W_i_i;   mode = 2; break;
    default: W = W_e_out; mode = 1; break;
  }
  float w = W[((M * CH + K) * 3 + ky) * 3 + kx];
  if (mode == 1)      w =  fmaxf(w, 0.f);
  else if (mode == 2) w = -fmaxf(w, 0.f);
  ws[gid] = (_Float16)w;
}

// ---------------------------------------------------------------------------
// Kernel 1: h_e_new / h_i_new. One block per (batch,row). 8 waves, each owns
// a 16-pixel segment. LDS stages 3 tensors x 3 rows x 130 cols x 32ch (f16,
// channel-last, zero halo). 108 v_wmma_f32_16x16x32_f16 per wave.
// ---------------------------------------------------------------------------
__global__ __launch_bounds__(256)
void rnn_state_kernel(const float* __restrict__ x,  const float* __restrict__ he,
                      const float* __restrict__ hi,
                      const float* __restrict__ b_in_e, const float* __restrict__ b_e_e,
                      const float* __restrict__ b_i_e,  const float* __restrict__ b_in_i,
                      const float* __restrict__ b_e_i,  const float* __restrict__ b_i_i,
                      const float* __restrict__ tau_e,  const float* __restrict__ tau_i,
                      const _Float16* __restrict__ wfrag,
                      float* __restrict__ he_new, float* __restrict__ hi_new) {
  __shared__ __align__(32) _Float16 smem[3 * 3 * COLS * CH];  // 74,880 B

  const int tid = threadIdx.x;
  const int b   = blockIdx.x >> 7;
  const int y   = blockIdx.x & 127;

  // Zero the halo columns (colp 0 and 129): 3 tensors * 3 rows * 2 sides * 2
  // 16-half chunks = 36 chunks of 32B.
  if (tid < 36) {
    int t    = tid / 12;
    int r    = (tid % 12) >> 2;
    int q    = tid & 3;
    int colp = (q & 2) ? (COLS - 1) : 0;
    int c0   = (q & 1) * 16;
    v16h z = {};
    *(v16h*)&smem[(((t * 3 + r) * COLS) + colp) * CH + c0] = z;
  }

  // Stage: thread -> (col = tid&127, channel-halves tid>>7), f32->f16,
  // channel-last so B-fragments are contiguous 32B LDS reads.
  {
    const float* srcs[3] = {x, he, hi};
    const int col = tid & 127;
    const int c0  = (tid >> 7) * 16;
#pragma unroll
    for (int t = 0; t < 3; ++t) {
      const float* src = srcs[t] + (size_t)b * PLANE;
#pragma unroll
      for (int r = 0; r < 3; ++r) {
        const int  yy    = y + r - 1;
        const bool valid = (yy >= 0) && (yy < HWD);
        v16h tmp;
        if (valid) {
          const float* p = src + (size_t)c0 * IMG + (size_t)yy * HWD + col;
#pragma unroll
          for (int i = 0; i < 16; ++i) tmp[i] = (_Float16)p[(size_t)i * IMG];
        } else {
#pragma unroll
          for (int i = 0; i < 16; ++i) tmp[i] = (_Float16)0.f;
        }
        *(v16h*)&smem[(((t * 3 + r) * COLS) + (col + 1)) * CH + c0] = tmp;
      }
    }
  }
  __syncthreads();

  const int lane  = tid & 31;
  const int seg   = (tid >> 5) * 16;           // wave's 16-pixel segment
  const int nIdx  = lane & 15;                 // N within tile
  const int cHalf = (lane >= 16) ? 16 : 0;     // B-matrix K-half select

  v8f accE0 = {}, accE1 = {}, accI0 = {}, accI1 = {};

#pragma unroll 1
  for (int t = 0; t < 3; ++t) {
#pragma unroll
    for (int tap = 0; tap < 9; ++tap) {
      const int ky = tap / 3, kx = tap % 3;
      // B fragment: 32x16 patch (K = input channels), one 32B LDS read.
      const v16h bfr = *(const v16h*)
          &smem[(((t * 3 + ky) * COLS) + (seg + nIdx + kx)) * CH + cHalf];
      // A fragments from repacked weights (L2-resident).
      const _Float16* aE = wfrag + (size_t)((t * 9 + tap) * 2) * 512 + lane * 16;
      const _Float16* aI = wfrag + (size_t)(((t + 3) * 9 + tap) * 2) * 512 + lane * 16;
      const v16h aE0 = *(const v16h*)(aE);
      const v16h aE1 = *(const v16h*)(aE + 512);
      const v16h aI0 = *(const v16h*)(aI);
      const v16h aI1 = *(const v16h*)(aI + 512);
      accE0 = __builtin_amdgcn_wmma_f32_16x16x32_f16(false, aE0, false, bfr, (short)0, accE0, false, false);
      accE1 = __builtin_amdgcn_wmma_f32_16x16x32_f16(false, aE1, false, bfr, (short)0, accE1, false, false);
      accI0 = __builtin_amdgcn_wmma_f32_16x16x32_f16(false, aI0, false, bfr, (short)0, accI0, false, false);
      accI1 = __builtin_amdgcn_wmma_f32_16x16x32_f16(false, aI1, false, bfr, (short)0, accI1, false, false);
    }
  }

  // Epilogue: bias (+relu / -relu), tanh, tau-leaky integration, store.
  const int col    = seg + nIdx;
  const int rowOff = (lane >= 16) ? 8 : 0;     // C/D layout: lanes 16-31 -> M+8
#pragma unroll
  for (int m = 0; m < 2; ++m) {
    const v8f aE = m ? accE1 : accE0;
    const v8f aI = m ? accI1 : accI0;
#pragma unroll
    for (int r = 0; r < 8; ++r) {
      const int    OC  = m * 16 + r + rowOff;
      const size_t idx = ((size_t)(b * CH + OC) * HWD + y) * HWD + col;

      const float biasE = b_in_e[OC] + fmaxf(b_e_e[OC], 0.f) - fmaxf(b_i_e[OC], 0.f);
      const float ce    = tanhf(aE[r] + biasE);
      const float invE  = 1.f / fmaxf(tau_e[OC], 1.f);
      he_new[idx] = (1.f - invE) * he[idx] + invE * ce;

      const float biasI = b_in_i[OC] + fmaxf(b_e_i[OC], 0.f) - fmaxf(b_i_i[OC], 0.f);
      const float ci    = tanhf(aI[r] + biasI);
      const float invI  = 1.f / fmaxf(tau_i[OC], 1.f);
      hi_new[idx] = (1.f - invI) * hi[idx] + invI * ci;
    }
  }
}

// ---------------------------------------------------------------------------
// Kernel 2: out = tanh(conv(h_e_new, relu(W_e_out)) + relu(b_e_out)).
// ---------------------------------------------------------------------------
__global__ __launch_bounds__(256)
void rnn_out_kernel(const float* __restrict__ he_new,
                    const float* __restrict__ b_e_out,
                    const _Float16* __restrict__ wfrag,
                    float* __restrict__ out) {
  __shared__ __align__(32) _Float16 smem[3 * COLS * CH];  // 24,960 B

  const int tid = threadIdx.x;
  const int b   = blockIdx.x >> 7;
  const int y   = blockIdx.x & 127;

  if (tid < 12) {  // halo zero: 3 rows * 2 sides * 2 chunks
    int r    = tid >> 2;
    int q    = tid & 3;
    int colp = (q & 2) ? (COLS - 1) : 0;
    int c0   = (q & 1) * 16;
    v16h z = {};
    *(v16h*)&smem[((r * COLS) + colp) * CH + c0] = z;
  }
  {
    const int col = tid & 127;
    const int c0  = (tid >> 7) * 16;
    const float* src = he_new + (size_t)b * PLANE;
#pragma unroll
    for (int r = 0; r < 3; ++r) {
      const int  yy    = y + r - 1;
      const bool valid = (yy >= 0) && (yy < HWD);
      v16h tmp;
      if (valid) {
        const float* p = src + (size_t)c0 * IMG + (size_t)yy * HWD + col;
#pragma unroll
        for (int i = 0; i < 16; ++i) tmp[i] = (_Float16)p[(size_t)i * IMG];
      } else {
#pragma unroll
        for (int i = 0; i < 16; ++i) tmp[i] = (_Float16)0.f;
      }
      *(v16h*)&smem[((r * COLS) + (col + 1)) * CH + c0] = tmp;
    }
  }
  __syncthreads();

  const int lane  = tid & 31;
  const int seg   = (tid >> 5) * 16;
  const int nIdx  = lane & 15;
  const int cHalf = (lane >= 16) ? 16 : 0;

  v8f acc0 = {}, acc1 = {};
#pragma unroll
  for (int tap = 0; tap < 9; ++tap) {
    const int ky = tap / 3, kx = tap % 3;
    const v16h bfr = *(const v16h*)
        &smem[((ky * COLS) + (seg + nIdx + kx)) * CH + cHalf];
    const _Float16* aO = wfrag + (size_t)((6 * 9 + tap) * 2) * 512 + lane * 16;
    const v16h a0 = *(const v16h*)(aO);
    const v16h a1 = *(const v16h*)(aO + 512);
    acc0 = __builtin_amdgcn_wmma_f32_16x16x32_f16(false, a0, false, bfr, (short)0, acc0, false, false);
    acc1 = __builtin_amdgcn_wmma_f32_16x16x32_f16(false, a1, false, bfr, (short)0, acc1, false, false);
  }

  const int col    = seg + nIdx;
  const int rowOff = (lane >= 16) ? 8 : 0;
#pragma unroll
  for (int m = 0; m < 2; ++m) {
    const v8f a = m ? acc1 : acc0;
#pragma unroll
    for (int r = 0; r < 8; ++r) {
      const int    OC  = m * 16 + r + rowOff;
      const size_t idx = ((size_t)(b * CH + OC) * HWD + y) * HWD + col;
      out[idx] = tanhf(a[r] + fmaxf(b_e_out[OC], 0.f));
    }
  }
}

extern "C" void kernel_launch(void* const* d_in, const int* in_sizes, int n_in,
                              void* d_out, int out_size, void* d_ws, size_t ws_size,
                              hipStream_t stream) {
  const float* x       = (const float*)d_in[0];
  const float* h_e     = (const float*)d_in[1];
  const float* h_i     = (const float*)d_in[2];
  const float* W_in_e  = (const float*)d_in[3];
  const float* b_in_e  = (const float*)d_in[4];
  const float* W_in_i  = (const float*)d_in[5];
  const float* b_in_i  = (const float*)d_in[6];
  const float* W_e_e   = (const float*)d_in[7];
  const float* b_e_e   = (const float*)d_in[8];
  const float* W_e_i   = (const float*)d_in[9];
  const float* b_e_i   = (const float*)d_in[10];
  const float* W_e_out = (const float*)d_in[11];
  const float* b_e_out = (const float*)d_in[12];
  const float* W_i_e   = (const float*)d_in[13];
  const float* b_i_e   = (const float*)d_in[14];
  const float* W_i_i   = (const float*)d_in[15];
  const float* b_i_i   = (const float*)d_in[16];
  const float* tau_e   = (const float*)d_in[17];
  const float* tau_i   = (const float*)d_in[18];

  float* out    = (float*)d_out;
  float* he_new = out + NPIX;
  float* hi_new = out + 2 * NPIX;
  _Float16* wfrag = (_Float16*)d_ws;  // 126 fragments * 1KB = 129,024 B

  repack_weights_kernel<<<252, 256, 0, stream>>>(
      W_in_e, W_e_e, W_i_e, W_in_i, W_e_i, W_i_i, W_e_out, wfrag);

  rnn_state_kernel<<<16 * 128, 256, 0, stream>>>(
      x, h_e, h_i, b_in_e, b_e_e, b_i_e, b_in_i, b_e_i, b_i_i,
      tau_e, tau_i, wfrag, he_new, hi_new);

  rnn_out_kernel<<<16 * 128, 256, 0, stream>>>(he_new, b_e_out, wfrag, out);
}
